// GumbelVectorQuantizer_Predictive_83124797047448
// MI455X (gfx1250) — compile-verified, matrix operand order checked by reading, not run on hardware
//
#include <hip/hip_runtime.h>

#define CTX 7
#define Bb 8
#define Tt 4096
#define Dd 256
#define Mm 1024
#define Ss (Tt + CTX - 1)   /* 4102 */
#define Nn (Bb * Tt)        /* 32768 */

typedef __attribute__((ext_vector_type(16))) __bf16        v16bf;
typedef __attribute__((ext_vector_type(8)))  float         v8f;
typedef __attribute__((ext_vector_type(4)))  unsigned int  uint4v;

union Frag { uint4v q[2]; v16bf bf; };

__device__ __forceinline__ unsigned short f2bf(float f) {
  unsigned int x = __float_as_uint(f);
  unsigned int r = x + 0x7FFFu + ((x >> 16) & 1u);   // round to nearest even
  return (unsigned short)(r >> 16);
}

__device__ __forceinline__ uint4v ld128(const unsigned short* p) {
  return *reinterpret_cast<const uint4v*>(p);
}

__device__ __forceinline__ v8f wmma_bf16(const Frag& A, const Frag& B, v8f C) {
  return __builtin_amdgcn_wmma_f32_16x16x32_bf16(false, A.bf, false, B.bf,
                                                 (short)0, C, false, false);
}

// ---------------------------------------------------------------- zero accums
__global__ void k_zero(float* scale_raw, float* counts, float* avg) {
  int t = threadIdx.x;
  if (t < Bb) scale_raw[t] = 0.f;
  counts[t] = 0.f;
  avg[t]    = 0.f;
}

// ------------------------------------------------- sum of squares for scale
__global__ void k_sumsq(const float* __restrict__ x, float* scale_raw) {
  __shared__ float red[256];
  int b = blockIdx.x >> 6;
  int chunk = blockIdx.x & 63;
  const int per = ((Tt - 1) * Dd) / 64;        // 16380
  const float* xb = x + (size_t)b * Tt * Dd + (size_t)chunk * per;
  float s = 0.f;
  for (int i = threadIdx.x; i < per; i += 256) { float v = xb[i]; s += v * v; }
  red[threadIdx.x] = s; __syncthreads();
  for (int o = 128; o > 0; o >>= 1) {
    if (threadIdx.x < o) red[threadIdx.x] += red[threadIdx.x + o];
    __syncthreads();
  }
  if (threadIdx.x == 0) atomicAdd(scale_raw + b, red[0]);
}

// --------------------------------------- embedding -> bf16 + |E_m|^2 biases
__global__ void k_e2(const float* __restrict__ E, unsigned short* E_bf, float* e2) {
  __shared__ float red[256];
  int m = blockIdx.x;
  float v = E[(size_t)m * Dd + threadIdx.x];
  E_bf[(size_t)m * Dd + threadIdx.x] = f2bf(v);
  red[threadIdx.x] = v * v; __syncthreads();
  for (int o = 128; o > 0; o >>= 1) {
    if (threadIdx.x < o) red[threadIdx.x] += red[threadIdx.x + o];
    __syncthreads();
  }
  if (threadIdx.x == 0) e2[m] = red[0];
}

// ------------------------------------- w_ctx (o,i,k) -> bf16 [k][o][i]
__global__ void k_wconv(const float* __restrict__ w, unsigned short* __restrict__ o) {
  int i  = threadIdx.x;
  int k  = blockIdx.x >> 8;
  int oc = blockIdx.x & 255;
  o[(((size_t)k * Dd) + oc) * Dd + i] = f2bf(w[((size_t)oc * Dd + i) * CTX + k]);
}

// ------------------------------------- w_f1/w_f2 -> bf16 (already [o][c])
__global__ void k_wf(const float* __restrict__ w1, const float* __restrict__ w2,
                     unsigned short* __restrict__ o1, unsigned short* __restrict__ o2) {
  int id = blockIdx.x * 256 + threadIdx.x;
  const int half = Dd * 2 * Dd;               // 131072
  if (id < half) o1[id] = f2bf(w1[id]);
  else           o2[id - half] = f2bf(w2[id - half]);
}

// ------------------------------------------------------- x -> bf16 rows
__global__ void k_xbf(const float* __restrict__ x, unsigned short* __restrict__ xb) {
  size_t id = (size_t)blockIdx.x * Dd + threadIdx.x;
  xb[id] = f2bf(x[id]);
}

// ----------------------- padded+noised context input -> bf16 [b][s][c]
__global__ void k_cin(const float* __restrict__ x, const float* __restrict__ noise,
                      const int* __restrict__ epo, const float* __restrict__ scale_raw,
                      unsigned short* __restrict__ cin_bf) {
  int c  = threadIdx.x;
  int s_ = blockIdx.x % Ss;
  int b  = blockIdx.x / Ss;
  float decay = exp2f(-(float)epo[0] * 0.1f);
  float sc = sqrtf(scale_raw[b] * (1.0f / ((float)Dd * (float)Ss)));
  float nf = 0.5f * sc * decay;
  float v = nf * noise[((size_t)b * Dd + c) * Ss + s_];
  if (s_ >= CTX) v += x[((size_t)b * Tt + (s_ - CTX)) * Dd + c];
  cin_bf[((size_t)b * Ss + s_) * Dd + c] = f2bf(v);
}

// -------------------------------- context conv as 7 shifted WMMA GEMMs
__global__ void k_conv(const unsigned short* __restrict__ cin_bf,
                       const unsigned short* __restrict__ wctx_bf,
                       unsigned short* __restrict__ ctxout_bf) {
  int lane = threadIdx.x & 31, w = threadIdx.x >> 5;
  int b  = blockIdx.z;
  int t0 = blockIdx.x * 64 + (w & 3) * 16;
  int o0 = blockIdx.y * 32 + (w >> 2) * 16;
  int l15 = lane & 15, kq = (lane >> 4) * 8;
  const unsigned short* aRow  = cin_bf + ((size_t)b * Ss + (t0 + l15)) * Dd;
  const unsigned short* bRow0 = wctx_bf + (size_t)(o0 + l15) * Dd;
  v8f acc = {0.f, 0.f, 0.f, 0.f, 0.f, 0.f, 0.f, 0.f};
  for (int k = 0; k < CTX; ++k) {
    const unsigned short* aR = aRow + (size_t)k * Dd;
    const unsigned short* bR = bRow0 + (size_t)k * (Dd * Dd);
    for (int kb = 0; kb < Dd; kb += 32) {
      Frag A, B;
      A.q[0] = ld128(aR + kb + kq);      A.q[1] = ld128(aR + kb + 16 + kq);
      B.q[0] = ld128(bR + kb + kq);      B.q[1] = ld128(bR + kb + 16 + kq);
      acc = wmma_bf16(A, B, acc);
    }
  }
  int rb = t0 + (lane >> 4) * 8;
  for (int r = 0; r < 8; ++r)
    ctxout_bf[((size_t)b * Tt + rb + r) * Dd + o0 + l15] = f2bf(acc[r]);
}

// -------------------- fuse1: cur = prelu(W1 [x; ctx], a1)  (K = 512)
__global__ void k_fuse1(const unsigned short* __restrict__ x_bf,
                        const unsigned short* __restrict__ ctxout_bf,
                        const unsigned short* __restrict__ wf1_bf,
                        const float* __restrict__ a1p,
                        unsigned short* __restrict__ cur_bf) {
  int lane = threadIdx.x & 31, w = threadIdx.x >> 5;
  int r0 = blockIdx.x * 64 + (w & 3) * 16;
  int o0 = blockIdx.y * 32 + (w >> 2) * 16;
  int l15 = lane & 15, kq = (lane >> 4) * 8;
  const unsigned short* aX = x_bf      + (size_t)(r0 + l15) * Dd;
  const unsigned short* aC = ctxout_bf + (size_t)(r0 + l15) * Dd;
  const unsigned short* bR = wf1_bf    + (size_t)(o0 + l15) * (2 * Dd);
  v8f acc = {0.f, 0.f, 0.f, 0.f, 0.f, 0.f, 0.f, 0.f};
  for (int c0 = 0; c0 < 2 * Dd; c0 += 32) {
    const unsigned short* aB = (c0 < Dd) ? (aX + c0) : (aC + (c0 - Dd));
    Frag A, B;
    A.q[0] = ld128(aB + kq);            A.q[1] = ld128(aB + 16 + kq);
    B.q[0] = ld128(bR + c0 + kq);       B.q[1] = ld128(bR + c0 + 16 + kq);
    acc = wmma_bf16(A, B, acc);
  }
  float a1 = a1p[0];
  int rb = r0 + (lane >> 4) * 8;
  for (int r = 0; r < 8; ++r) {
    float v = acc[r];
    v = (v >= 0.f) ? v : a1 * v;
    cur_bf[(size_t)(rb + r) * Dd + o0 + l15] = f2bf(v);
  }
}

// -------- distances + hard argmax + gumbel argmax + softmax col-sums
__global__ void k_dist(const unsigned short* __restrict__ cur_bf,
                       const unsigned short* __restrict__ E_bf,
                       const float* __restrict__ e2,
                       const float* __restrict__ gumbel_u,
                       float* __restrict__ counts,
                       float* __restrict__ avg,
                       int* __restrict__ idx_ws,
                       float* __restrict__ out_inds) {
  __shared__ float sVal[8][16];
  __shared__ int   sIdx[8][16];
  __shared__ float rowM[16];
  __shared__ float rowInv[16];
  int lane = threadIdx.x & 31, w = threadIdx.x >> 5;
  int rowbase = blockIdx.x * 16;
  int l15 = lane & 15, half = lane >> 4, kq = half * 8;
  int m0 = w * 128;

  // A fragments for these 16 rows (K = 256)
  Frag Af[8];
  const unsigned short* aR = cur_bf + (size_t)(rowbase + l15) * Dd;
  for (int kb = 0; kb < 8; ++kb) {
    Af[kb].q[0] = ld128(aR + kb * 32 + kq);
    Af[kb].q[1] = ld128(aR + kb * 32 + 16 + kq);
  }
  float sv[8][8];   // [tile j][vgpr r]: s_m = 2 cur.E - |E|^2
  for (int j = 0; j < 8; ++j) {
    int mc = m0 + j * 16 + l15;
    const unsigned short* bR = E_bf + (size_t)mc * Dd;
    v8f acc = {0.f, 0.f, 0.f, 0.f, 0.f, 0.f, 0.f, 0.f};
    for (int kb = 0; kb < 8; ++kb) {
      Frag B;
      B.q[0] = ld128(bR + kb * 32 + kq);
      B.q[1] = ld128(bR + kb * 32 + 16 + kq);
      acc = wmma_bf16(Af[kb], B, acc);
    }
    float bias = e2[mc];
    for (int r = 0; r < 8; ++r) sv[j][r] = 2.0f * acc[r] - bias;
  }

  // phase A: hard argmax per row -> code usage counts, row max for softmax
  for (int r = 0; r < 8; ++r) {
    float bv = -3.4e38f; int bm = 0;
    for (int j = 0; j < 8; ++j) {
      float v = sv[j][r]; int m_ = m0 + j * 16 + l15;
      if (v > bv) { bv = v; bm = m_; }
    }
    for (int msk = 1; msk < 16; msk <<= 1) {
      float ov = __shfl_xor(bv, msk, 32);
      int   om = __shfl_xor(bm, msk, 32);
      if (ov > bv || (ov == bv && om < bm)) { bv = ov; bm = om; }
    }
    if (l15 == 0) { int row = half * 8 + r; sVal[w][row] = bv; sIdx[w][row] = bm; }
  }
  __syncthreads();
  if (threadIdx.x < 16) {
    int row = threadIdx.x;
    float bv = sVal[0][row]; int bm = sIdx[0][row];
    for (int ww = 1; ww < 8; ++ww) {
      float ov = sVal[ww][row]; int om = sIdx[ww][row];
      if (ov > bv || (ov == bv && om < bm)) { bv = ov; bm = om; }
    }
    atomicAdd(counts + bm, 1.0f);
    rowM[row] = bv;
  }
  __syncthreads();

  // phase B: gumbel-perturbed argmax -> quantization indices
  for (int r = 0; r < 8; ++r) {
    int row = half * 8 + r;
    const float* gRow = gumbel_u + (size_t)(rowbase + row) * Mm;
    float bv = -3.4e38f; int bm = 0;
    for (int j = 0; j < 8; ++j) {
      int m_ = m0 + j * 16 + l15;
      float u = gRow[m_];
      u = fminf(fmaxf(u, 1e-10f), 1.0f - 1e-7f);
      float g = -__logf(-__logf(u));
      float v = sv[j][r] + g;
      if (v > bv) { bv = v; bm = m_; }
    }
    for (int msk = 1; msk < 16; msk <<= 1) {
      float ov = __shfl_xor(bv, msk, 32);
      int   om = __shfl_xor(bm, msk, 32);
      if (ov > bv || (ov == bv && om < bm)) { bv = ov; bm = om; }
    }
    if (l15 == 0) { sVal[w][row] = bv; sIdx[w][row] = bm; }
  }
  __syncthreads();
  if (threadIdx.x < 16) {
    int row = threadIdx.x;
    float bv = sVal[0][row]; int bm = sIdx[0][row];
    for (int ww = 1; ww < 8; ++ww) {
      float ov = sVal[ww][row]; int om = sIdx[ww][row];
      if (ov > bv || (ov == bv && om < bm)) { bv = ov; bm = om; }
    }
    idx_ws[rowbase + row] = bm;
    out_inds[rowbase + row] = (float)bm;
  }
  __syncthreads();

  // phase C: softmax column sums -> avg probs
  for (int r = 0; r < 8; ++r) {
    int row = half * 8 + r;
    float mx = rowM[row];
    float s = 0.f;
    for (int j = 0; j < 8; ++j) { float e = __expf(sv[j][r] - mx); sv[j][r] = e; s += e; }
    for (int msk = 1; msk < 16; msk <<= 1) s += __shfl_xor(s, msk, 32);
    if (l15 == 0) sVal[w][row] = s;
  }
  __syncthreads();
  if (threadIdx.x < 16) {
    int row = threadIdx.x;
    float z = 0.f;
    for (int ww = 0; ww < 8; ++ww) z += sVal[ww][row];
    rowInv[row] = 1.0f / z;
  }
  __syncthreads();
  for (int j = 0; j < 8; ++j) {
    float cp = 0.f;
    for (int r = 0; r < 8; ++r) { int row = half * 8 + r; cp += sv[j][r] * rowInv[row]; }
    cp += __shfl_xor(cp, 16, 32);
    if (half == 0) atomicAdd(avg + m0 + j * 16 + l15, cp);
  }
}

// ---- fuse2: quantized = prelu(W2 [E[idx]; ctx], a2); hard ST -> gather
__global__ void k_fuse2(const unsigned short* __restrict__ E_bf,
                        const unsigned short* __restrict__ ctxout_bf,
                        const unsigned short* __restrict__ wf2_bf,
                        const int* __restrict__ idx_ws,
                        const float* __restrict__ a2p,
                        float* __restrict__ out_q) {
  int lane = threadIdx.x & 31, w = threadIdx.x >> 5;
  int r0 = blockIdx.x * 64 + (w & 3) * 16;
  int o0 = blockIdx.y * 32 + (w >> 2) * 16;
  int l15 = lane & 15, kq = (lane >> 4) * 8;
  int myRow = r0 + l15;
  const unsigned short* aQ = E_bf      + (size_t)idx_ws[myRow] * Dd;
  const unsigned short* aC = ctxout_bf + (size_t)myRow * Dd;
  const unsigned short* bR = wf2_bf    + (size_t)(o0 + l15) * (2 * Dd);
  v8f acc = {0.f, 0.f, 0.f, 0.f, 0.f, 0.f, 0.f, 0.f};
  for (int c0 = 0; c0 < 2 * Dd; c0 += 32) {
    const unsigned short* aB = (c0 < Dd) ? (aQ + c0) : (aC + (c0 - Dd));
    Frag A, B;
    A.q[0] = ld128(aB + kq);            A.q[1] = ld128(aB + 16 + kq);
    B.q[0] = ld128(bR + c0 + kq);       B.q[1] = ld128(bR + c0 + 16 + kq);
    acc = wmma_bf16(A, B, acc);
  }
  float a2 = a2p[0];
  int rb = r0 + (lane >> 4) * 8;
  for (int r = 0; r < 8; ++r) {
    float v = acc[r];
    v = (v >= 0.f) ? v : a2 * v;
    out_q[(size_t)(rb + r) * Dd + o0 + l15] = v;
  }
}

// ---------------------------------------------------- entropies
__global__ void k_final(const float* __restrict__ counts, const float* __restrict__ avg,
                        float* __restrict__ out2) {
  __shared__ float redH[1024];
  __shared__ float redS[1024];
  int t = threadIdx.x;
  float ph = counts[t] * (1.0f / (float)Nn);
  float ps = avg[t]    * (1.0f / (float)Nn);
  redH[t] = -ph * log2f(ph + 1e-10f);
  redS[t] = -ps * log2f(ps + 1e-10f);
  __syncthreads();
  for (int o = 512; o > 0; o >>= 1) {
    if (t < o) { redH[t] += redH[t + o]; redS[t] += redS[t + o]; }
    __syncthreads();
  }
  if (t == 0) { out2[0] = redH[0]; out2[1] = redS[0]; }
}

extern "C" void kernel_launch(void* const* d_in, const int* in_sizes, int n_in,
                              void* d_out, int out_size, void* d_ws, size_t ws_size,
                              hipStream_t stream) {
  (void)in_sizes; (void)n_in; (void)out_size; (void)ws_size;
  const float* x      = (const float*)d_in[0];
  const float* noise  = (const float*)d_in[1];
  const float* gumbel = (const float*)d_in[2];
  const int*   epo    = (const int*)  d_in[3];
  const float* embed  = (const float*)d_in[4];
  const float* w_ctx  = (const float*)d_in[5];
  const float* w_f1   = (const float*)d_in[6];
  const float* a1     = (const float*)d_in[7];
  const float* w_f2   = (const float*)d_in[8];
  const float* a2     = (const float*)d_in[9];
  float* out = (float*)d_out;

  char* wsp = (char*)d_ws;
  size_t off = 0;
  auto alloc = [&](size_t bytes) -> char* {
    char* p = wsp + off;
    off = (off + bytes + 255) & ~(size_t)255;
    return p;
  };
  float* scale_raw = (float*)alloc(Bb * 4);
  float* counts    = (float*)alloc(Mm * 4);
  float* avg       = (float*)alloc(Mm * 4);
  float* e2        = (float*)alloc(Mm * 4);
  int*   idx_ws    = (int*)  alloc((size_t)Nn * 4);
  unsigned short* x_bf      = (unsigned short*)alloc((size_t)Nn * Dd * 2);
  unsigned short* cin_bf    = (unsigned short*)alloc((size_t)Bb * Ss * Dd * 2);
  unsigned short* ctxout_bf = (unsigned short*)alloc((size_t)Nn * Dd * 2);
  unsigned short* cur_bf    = (unsigned short*)alloc((size_t)Nn * Dd * 2);
  unsigned short* wctx_bf   = (unsigned short*)alloc((size_t)CTX * Dd * Dd * 2);
  unsigned short* wf1_bf    = (unsigned short*)alloc((size_t)Dd * 2 * Dd * 2);
  unsigned short* wf2_bf    = (unsigned short*)alloc((size_t)Dd * 2 * Dd * 2);
  unsigned short* E_bf      = (unsigned short*)alloc((size_t)Mm * Dd * 2);

  float* out_q    = out;                        // [B*T, D]
  float* out_perp = out + (size_t)Nn * Dd;      // 2 scalars
  float* out_inds = out_perp + 2;               // [B*T]

  k_zero  <<<1, 1024, 0, stream>>>(scale_raw, counts, avg);
  k_sumsq <<<Bb * 64, 256, 0, stream>>>(x, scale_raw);
  k_e2    <<<Mm, 256, 0, stream>>>(embed, E_bf, e2);
  k_wconv <<<CTX * Dd, 256, 0, stream>>>(w_ctx, wctx_bf);
  k_wf    <<<1024, 256, 0, stream>>>(w_f1, w_f2, wf1_bf, wf2_bf);
  k_xbf   <<<Nn, 256, 0, stream>>>(x, x_bf);
  k_cin   <<<Bb * Ss, 256, 0, stream>>>(x, noise, epo, scale_raw, cin_bf);
  k_conv  <<<dim3(Tt / 64, Dd / 32, Bb), 256, 0, stream>>>(cin_bf, wctx_bf, ctxout_bf);
  k_fuse1 <<<dim3(Nn / 64, Dd / 32), 256, 0, stream>>>(x_bf, ctxout_bf, wf1_bf, a1, cur_bf);
  k_dist  <<<Nn / 16, 256, 0, stream>>>(cur_bf, E_bf, e2, gumbel, counts, avg, idx_ws, out_inds);
  k_fuse2 <<<dim3(Nn / 64, Dd / 32), 256, 0, stream>>>(E_bf, ctxout_bf, wf2_bf, idx_ws, a2, out_q);
  k_final <<<1, 1024, 0, stream>>>(counts, avg, out_perp);
}
